// PointNetPlusPlus_68719477565
// MI455X (gfx1250) — matrix operand order repaired and checked by reading, not compiled, not run on hardware
//
#include <hip/hip_runtime.h>

#define DEVFN __device__ __forceinline__

typedef __attribute__((ext_vector_type(8)))  float  v8f;
typedef __attribute__((ext_vector_type(8)))  __bf16 v8bf;
typedef __attribute__((ext_vector_type(16))) __bf16 v16bf;

static constexpr int BB   = 8;      // batches
static constexpr int NPTS = 8192;   // points per cloud
static constexpr int M1   = 1639;   // SA1 centroids
static constexpr int M2   = 410;    // SA2 centroids
static constexpr int KNBR = 64;     // neighbors per centroid

// ---------------- helpers ----------------

// native f32 -> bf16 (hardware convert, round-to-nearest-even)
DEVFN unsigned short f32_bf16(float f) {
  return __builtin_bit_cast(unsigned short, (__bf16)f);
}

// order-preserving float -> uint encoding (for atomicMax-based pooling)
DEVFN unsigned enc_f32(float f) {
  unsigned u = __float_as_uint(f);
  return (u & 0x80000000u) ? ~u : (u | 0x80000000u);
}
DEVFN float dec_f32(unsigned e) {
  unsigned u = (e & 0x80000000u) ? (e ^ 0x80000000u) : ~e;
  return __uint_as_float(u);
}

// A fragment (16x32 bf16) from row-major LDS, pitch in elements.
// ISA layout: lane<16: row=lane, e<8 -> K=e, e>=8 -> K=e+8
//             lane>=16: row=lane-16, K shifted by +8
DEVFN v16bf load_a_frag(const unsigned short* h, int pitch, int rt, int kt, int lane) {
  int row = rt * 16 + (lane & 15);
  const unsigned short* p = h + row * pitch + kt * 32 + ((lane >> 4) << 3);
  union { v16bf v; v8bf h8[2]; } u;
  u.h8[0] = *(const v8bf*)(p);
  u.h8[1] = *(const v8bf*)(p + 16);
  return u.v;
}

// B fragment (32x16 bf16) from pre-packed weights: fragment f = kt*NT+nt,
// per-lane 16 contiguous bf16 (coalesced 32B per lane).
DEVFN v16bf load_b_frag(const unsigned short* w, int NT, int kt, int nt, int lane) {
  const unsigned short* p = w + (((size_t)(kt * NT + nt) * 32 + (size_t)lane) << 4);
  union { v16bf v; v8bf h8[2]; } u;
  u.h8[0] = *(const v8bf*)(p);
  u.h8[1] = *(const v8bf*)(p + 8);
  return u.v;
}

DEVFN v8f wmma_bf16(v16bf a, v16bf b, v8f c) {
  return __builtin_amdgcn_wmma_f32_16x16x32_bf16(false, a, false, b, (short)0, c, false, false);
}

// ---------------- weight packing ----------------
// W [Cin, Cout] f32 row-major -> bf16 fragments [(kt*NT+nt)*32+lane][16]
// B layout: K = kt*32 + 8*(lane>>4) + (e&7) + 16*(e>>3) ; N = nt*16+(lane&15)
__global__ void pack_w_kernel(const float* __restrict__ W, unsigned short* __restrict__ out,
                              int Cin, int Cout, int KT, int NT) {
  int total = KT * NT * 512;
  int idx = blockIdx.x * 256 + threadIdx.x;
  if (idx >= total) return;
  int e    = idx & 15;
  int lane = (idx >> 4) & 31;
  int f    = idx >> 9;
  int nt = f % NT, kt = f / NT;
  int k  = kt * 32 + (e & 7) + ((e >> 3) << 4) + ((lane >> 4) << 3);
  int nn = nt * 16 + (lane & 15);
  float v = (k < Cin) ? W[(size_t)k * Cout + nn] : 0.f;
  out[idx] = f32_bf16(v);
}

// ---------------- farthest point sampling ----------------
// one workgroup per batch; dists in LDS; sequential argmax sweeps
__global__ void __launch_bounds__(1024) fps_kernel(const float* __restrict__ pos, int ps,
                                                   int n, int m,
                                                   int* __restrict__ out_idx,
                                                   float* __restrict__ out_pos) {
  __shared__ float dists[NPTS];
  __shared__ float red_v[32];
  __shared__ int   red_i[32];
  __shared__ float cur[3];
  int tid = threadIdx.x, lane = tid & 31, wave = tid >> 5;
  int b = blockIdx.x;
  const float* base = pos + (size_t)b * n * ps;
  for (int i = tid; i < n; i += 1024) dists[i] = __builtin_inff();
  if (tid == 0) {
    out_idx[(size_t)b * m] = 0;
    cur[0] = base[0]; cur[1] = base[1]; cur[2] = base[2];
    out_pos[(size_t)b * m * 3 + 0] = cur[0];
    out_pos[(size_t)b * m * 3 + 1] = cur[1];
    out_pos[(size_t)b * m * 3 + 2] = cur[2];
  }
  __syncthreads();
  for (int it = 1; it < m; ++it) {
    float cx = cur[0], cy = cur[1], cz = cur[2];
    float bv = -1.f; int bi = 0x7fffffff;
    for (int i = tid; i < n; i += 1024) {
      float dx = base[(size_t)i * ps + 0] - cx;
      float dy = base[(size_t)i * ps + 1] - cy;
      float dz = base[(size_t)i * ps + 2] - cz;
      float d  = dx * dx + dy * dy + dz * dz;
      float dm = fminf(dists[i], d);
      dists[i] = dm;
      if (dm > bv || (dm == bv && i < bi)) { bv = dm; bi = i; }
    }
#pragma unroll
    for (int off = 16; off > 0; off >>= 1) {
      float ov = __shfl_xor(bv, off);
      int   oi = __shfl_xor(bi, off);
      if (ov > bv || (ov == bv && oi < bi)) { bv = ov; bi = oi; }
    }
    if (lane == 0) { red_v[wave] = bv; red_i[wave] = bi; }
    __syncthreads();
    if (tid == 0) {
      float fv = red_v[0]; int fi = red_i[0];
      for (int w = 1; w < 32; ++w)
        if (red_v[w] > fv || (red_v[w] == fv && red_i[w] < fi)) { fv = red_v[w]; fi = red_i[w]; }
      out_idx[(size_t)b * m + it] = fi;
      float px = base[(size_t)fi * ps + 0];
      float py = base[(size_t)fi * ps + 1];
      float pz = base[(size_t)fi * ps + 2];
      cur[0] = px; cur[1] = py; cur[2] = pz;
      out_pos[((size_t)b * m + it) * 3 + 0] = px;
      out_pos[((size_t)b * m + it) * 3 + 1] = py;
      out_pos[((size_t)b * m + it) * 3 + 2] = pz;
    }
    __syncthreads();
  }
}

// ---------------- ball query (k-nearest within radius) ----------------
// one thread per centroid; per-thread sorted top-64 list in LDS (stride-128 layout)
__global__ void __launch_bounds__(128) ball_query_kernel(const float* __restrict__ pos, int ps,
                                                         const float* __restrict__ pos_s,
                                                         int n, int m, float r2,
                                                         int* __restrict__ nbr,
                                                         int* __restrict__ msk) {
  extern __shared__ unsigned char bqsm[];
  float* d2l = (float*)bqsm;                          // [KNBR][128]
  int*   idl = (int*)(bqsm + (size_t)KNBR * 128 * 4); // [KNBR][128]
  int tid = threadIdx.x;
  int c = blockIdx.x * 128 + tid;
  int b = blockIdx.y;
  bool active = c < m;
  size_t cg = (size_t)b * m + (active ? c : 0);
  float qx = pos_s[cg * 3 + 0], qy = pos_s[cg * 3 + 1], qz = pos_s[cg * 3 + 2];
  const float* base = pos + (size_t)b * n * ps;
  int cnt = 0;
  for (int i = 0; i < n; ++i) {                       // lockstep -> broadcast loads
    float dx = base[(size_t)i * ps + 0] - qx;
    float dy = base[(size_t)i * ps + 1] - qy;
    float dz = base[(size_t)i * ps + 2] - qz;
    float d2 = dx * dx + dy * dy + dz * dz;
    if (!active || d2 > r2) continue;
    if (cnt == KNBR && d2 >= d2l[(KNBR - 1) * 128 + tid]) continue;
    int j = (cnt < KNBR) ? cnt : (KNBR - 1);
    while (j > 0 && d2l[(j - 1) * 128 + tid] > d2) {
      d2l[j * 128 + tid] = d2l[(j - 1) * 128 + tid];
      idl[j * 128 + tid] = idl[(j - 1) * 128 + tid];
      --j;
    }
    d2l[j * 128 + tid] = d2;
    idl[j * 128 + tid] = i;
    if (cnt < KNBR) ++cnt;
  }
  if (active) {
    int pad = (cnt > 0) ? idl[tid] : 0;
    for (int j = 0; j < KNBR; ++j) {
      nbr[cg * KNBR + j] = (j < cnt) ? idl[j * 128 + tid] : pad;
      msk[cg * KNBR + j] = (j < cnt) ? 1 : 0;
    }
  }
}

// ---------------- WMMA hidden layer: hin[ROWS][CIN] -> relu -> hout[ROWS][COUT] ----------------
template <int ROWS, int CIN, int COUT, bool RELU>
DEVFN void gemm_layer(const unsigned short* hin, unsigned short* hout,
                      const unsigned short* W, const float* bias, int wave, int lane) {
  constexpr int NT = COUT / 16, KT = CIN / 32, RT = ROWS / 16;
  for (int t = wave; t < RT * NT; t += 8) {
    int rt = t / NT, nt = t % NT;
    v8f acc = {0.f, 0.f, 0.f, 0.f, 0.f, 0.f, 0.f, 0.f};
#pragma unroll
    for (int kt = 0; kt < KT; ++kt) {
      v16bf a = load_a_frag(hin, CIN, rt, kt, lane);
      v16bf b = load_b_frag(W, NT, kt, nt, lane);
      acc = wmma_bf16(a, b, acc);
    }
    int col = nt * 16 + (lane & 15);
    float bv = bias[col];
    int rbase = rt * 16 + ((lane >> 4) << 3);
#pragma unroll
    for (int r = 0; r < 8; ++r) {
      float v = acc[r] + bv;
      if (RELU) v = fmaxf(v, 0.f);
      hout[(rbase + r) * COUT + col] = f32_bf16(v);
    }
  }
}

// ---------------- SA module conv: gather -> MLP(3 layers, WMMA) -> masked max pool ----------------
// GLOBAL=false: one workgroup per centroid (ROWS=64 neighbors), writes out[b][c][C3]
// GLOBAL=true : one workgroup per ROWS-chunk of rows, pools via encoded atomicMax into out_enc
template <int ROWS, int C0PAD, int CINF, int C1, int C2, int C3, bool GLOBAL>
__global__ void __launch_bounds__(256) sa_conv_kernel(
    const float* __restrict__ feat, int fs,
    const float* __restrict__ pos, int ps,
    const float* __restrict__ pos_s,
    const int* __restrict__ nbr, const int* __restrict__ msk,
    const unsigned short* __restrict__ W1, const float* __restrict__ b1,
    const unsigned short* __restrict__ W2, const float* __restrict__ b2,
    const unsigned short* __restrict__ W3, const float* __restrict__ b3,
    float* __restrict__ out, unsigned int* __restrict__ out_enc,
    int n, int m) {
  extern __shared__ unsigned short smem[];
  constexpr int PX = (C0PAD > C2) ? C0PAD : C2;
  unsigned short* hX = smem;                 // ROWS x PX  (layer0 input, layer2 output)
  unsigned short* hY = smem + ROWS * PX;     // ROWS x C1  (layer1 output)
  float* maskf = (float*)(smem + ROWS * PX + ROWS * C1);
  int tid = threadIdx.x, lane = tid & 31, wave = tid >> 5;
  int b = blockIdx.y;

  // ---- gather into hX (pitch C0PAD, bf16), fill maskf ----
  if (GLOBAL) {
    int g0 = blockIdx.x * ROWS;
    for (int j = wave; j < ROWS; j += 8) {
      int g = g0 + j;
      bool valid = g < n;
      int gg = valid ? g : 0;
      const float* fsrc = feat + ((size_t)b * n + gg) * fs;
      const float* psrc = pos + ((size_t)b * n + gg) * ps;
      for (int c = lane; c < C0PAD; c += 32) {
        float v = 0.f;
        if (valid) {
          if (c < CINF) v = fsrc[c];
          else if (c < CINF + 3) v = psrc[c - CINF];
        }
        hX[j * C0PAD + c] = f32_bf16(v);
      }
      if (lane == 0) maskf[j] = valid ? 1.f : 0.f;
    }
  } else {
    size_t cg = (size_t)b * m + blockIdx.x;
    float q0 = pos_s[cg * 3 + 0], q1 = pos_s[cg * 3 + 1], q2 = pos_s[cg * 3 + 2];
    for (int j = wave; j < ROWS; j += 8) {
      int p  = nbr[cg * KNBR + j];
      int mk = msk[cg * KNBR + j];
      const float* fsrc = feat + ((size_t)b * n + p) * fs;
      const float* psrc = pos + ((size_t)b * n + p) * ps;
      for (int c = lane; c < C0PAD; c += 32) {
        float v = 0.f;
        if (c < CINF) v = fsrc[c];
        else if (c == CINF)     v = psrc[0] - q0;
        else if (c == CINF + 1) v = psrc[1] - q1;
        else if (c == CINF + 2) v = psrc[2] - q2;
        hX[j * C0PAD + c] = f32_bf16(v);
      }
      if (lane == 0) maskf[j] = mk ? 1.f : 0.f;
    }
  }
  __syncthreads();

  gemm_layer<ROWS, C0PAD, C1, true>(hX, hY, W1, b1, wave, lane);
  __syncthreads();
  gemm_layer<ROWS, C1, C2, true>(hY, hX, W2, b2, wave, lane);
  __syncthreads();

  // ---- final layer fused with masked max over ROWS ----
  constexpr int NT3 = C3 / 16, KT3 = C2 / 32, RT = ROWS / 16;
  for (int nt = wave; nt < NT3; nt += 8) {
    float colmax = -__builtin_inff();
    for (int rt = 0; rt < RT; ++rt) {
      v8f acc = {0.f, 0.f, 0.f, 0.f, 0.f, 0.f, 0.f, 0.f};
#pragma unroll
      for (int kt = 0; kt < KT3; ++kt) {
        v16bf a = load_a_frag(hX, C2, rt, kt, lane);
        v16bf bm = load_b_frag(W3, NT3, kt, nt, lane);
        acc = wmma_bf16(a, bm, acc);
      }
      int rbase = rt * 16 + ((lane >> 4) << 3);
#pragma unroll
      for (int r = 0; r < 8; ++r) {
        float v = (maskf[rbase + r] > 0.5f) ? acc[r] : -__builtin_inff();
        colmax = fmaxf(colmax, v);
      }
    }
    // column n lives in lanes n and n+16 -> combine halves
    colmax = fmaxf(colmax, __shfl_xor(colmax, 16));
    if (lane < 16) {
      int col = nt * 16 + lane;
      float v = colmax + b3[col];
      if (GLOBAL) atomicMax(&out_enc[(size_t)b * C3 + col], enc_f32(v));
      else        out[((size_t)b * m + blockIdx.x) * C3 + col] = v;
    }
  }
}

// ---------------- misc ----------------
__global__ void init_genc_kernel(unsigned int* __restrict__ g, int count) {
  int i = blockIdx.x * 256 + threadIdx.x;
  if (i < count) g[i] = enc_f32(-__builtin_inff());
}

// head MLP 1024->512->256->10 on the 8 pooled vectors + log_softmax + broadcast to all N rows
__global__ void __launch_bounds__(512) head_kernel(const unsigned int* __restrict__ genc,
                                                   const float* __restrict__ W1, const float* __restrict__ b1,
                                                   const float* __restrict__ W2, const float* __restrict__ b2,
                                                   const float* __restrict__ W3, const float* __restrict__ b3,
                                                   float* __restrict__ out) {
  __shared__ float g[1024];
  __shared__ float t1[512];
  __shared__ float t2[256];
  __shared__ float o10[10];
  int t = threadIdx.x, b = blockIdx.x;
  for (int k = t; k < 1024; k += 512) g[k] = dec_f32(genc[(size_t)b * 1024 + k]);
  __syncthreads();
  {
    float acc = b1[t];
    for (int k = 0; k < 1024; ++k) acc += g[k] * W1[(size_t)k * 512 + t];
    t1[t] = fmaxf(acc, 0.f);
  }
  __syncthreads();
  if (t < 256) {
    float acc = b2[t];
    for (int k = 0; k < 512; ++k) acc += t1[k] * W2[(size_t)k * 256 + t];
    t2[t] = fmaxf(acc, 0.f);
  }
  __syncthreads();
  if (t < 10) {
    float acc = b3[t];
    for (int k = 0; k < 256; ++k) acc += t2[k] * W3[(size_t)k * 10 + t];
    o10[t] = acc;
  }
  __syncthreads();
  if (t == 0) {
    float mx = o10[0];
    for (int c = 1; c < 10; ++c) mx = fmaxf(mx, o10[c]);
    float s = 0.f;
    for (int c = 0; c < 10; ++c) s += expf(o10[c] - mx);
    float lse = mx + logf(s);
    for (int c = 0; c < 10; ++c) o10[c] -= lse;
  }
  __syncthreads();
  for (int row = t; row < NPTS; row += 512) {
    float* dst = out + ((size_t)b * NPTS + row) * 10;
#pragma unroll
    for (int c = 0; c < 10; ++c) dst[c] = o10[c];
  }
}

// ---------------- host launch ----------------
extern "C" void kernel_launch(void* const* d_in, const int* in_sizes, int n_in,
                              void* d_out, int out_size, void* d_ws, size_t ws_size,
                              hipStream_t stream) {
  (void)in_sizes; (void)n_in; (void)out_size; (void)ws_size;
  const float* x = (const float*)d_in[0];
  const float* s1W1 = (const float*)d_in[2];  const float* s1b1 = (const float*)d_in[3];
  const float* s1W2 = (const float*)d_in[4];  const float* s1b2 = (const float*)d_in[5];
  const float* s1W3 = (const float*)d_in[6];  const float* s1b3 = (const float*)d_in[7];
  const float* s2W1 = (const float*)d_in[8];  const float* s2b1 = (const float*)d_in[9];
  const float* s2W2 = (const float*)d_in[10]; const float* s2b2 = (const float*)d_in[11];
  const float* s2W3 = (const float*)d_in[12]; const float* s2b3 = (const float*)d_in[13];
  const float* s3W1 = (const float*)d_in[14]; const float* s3b1 = (const float*)d_in[15];
  const float* s3W2 = (const float*)d_in[16]; const float* s3b2 = (const float*)d_in[17];
  const float* s3W3 = (const float*)d_in[18]; const float* s3b3 = (const float*)d_in[19];
  const float* hW1  = (const float*)d_in[20]; const float* hb1  = (const float*)d_in[21];
  const float* hW2  = (const float*)d_in[22]; const float* hb2  = (const float*)d_in[23];
  const float* hW3  = (const float*)d_in[24]; const float* hb3  = (const float*)d_in[25];

  char* ws = (char*)d_ws;
  size_t off = 0;
  auto take = [&](size_t bytes) -> char* {
    char* p = ws + off;
    off += (bytes + 255) & ~(size_t)255;
    return p;
  };
  int*            idx1 = (int*)take((size_t)BB * M1 * 4);
  float*          ps1  = (float*)take((size_t)BB * M1 * 3 * 4);
  int*            nbr1 = (int*)take((size_t)BB * M1 * KNBR * 4);
  int*            msk1 = (int*)take((size_t)BB * M1 * KNBR * 4);
  float*          x1   = (float*)take((size_t)BB * M1 * 256 * 4);
  int*            idx2 = (int*)take((size_t)BB * M2 * 4);
  float*          ps2  = (float*)take((size_t)BB * M2 * 3 * 4);
  int*            nbr2 = (int*)take((size_t)BB * M2 * KNBR * 4);
  int*            msk2 = (int*)take((size_t)BB * M2 * KNBR * 4);
  float*          x2   = (float*)take((size_t)BB * M2 * 512 * 4);
  unsigned*       genc = (unsigned*)take((size_t)BB * 1024 * 4);
  unsigned short* P11  = (unsigned short*)take((size_t)1  * 8  * 512 * 2);
  unsigned short* P12  = (unsigned short*)take((size_t)4  * 8  * 512 * 2);
  unsigned short* P13  = (unsigned short*)take((size_t)4  * 16 * 512 * 2);
  unsigned short* P21  = (unsigned short*)take((size_t)9  * 16 * 512 * 2);
  unsigned short* P22  = (unsigned short*)take((size_t)8  * 16 * 512 * 2);
  unsigned short* P23  = (unsigned short*)take((size_t)8  * 32 * 512 * 2);
  unsigned short* P31  = (unsigned short*)take((size_t)17 * 32 * 512 * 2);
  unsigned short* P32  = (unsigned short*)take((size_t)16 * 32 * 512 * 2);
  unsigned short* P33  = (unsigned short*)take((size_t)16 * 64 * 512 * 2);

  auto packGrid = [](int KT, int NT) { return dim3((unsigned)((KT * NT * 512 + 255) / 256)); };
  pack_w_kernel<<<packGrid(1, 8),   256, 0, stream>>>(s1W1, P11, 6,   128,  1,  8);
  pack_w_kernel<<<packGrid(4, 8),   256, 0, stream>>>(s1W2, P12, 128, 128,  4,  8);
  pack_w_kernel<<<packGrid(4, 16),  256, 0, stream>>>(s1W3, P13, 128, 256,  4,  16);
  pack_w_kernel<<<packGrid(9, 16),  256, 0, stream>>>(s2W1, P21, 259, 256,  9,  16);
  pack_w_kernel<<<packGrid(8, 16),  256, 0, stream>>>(s2W2, P22, 256, 256,  8,  16);
  pack_w_kernel<<<packGrid(8, 32),  256, 0, stream>>>(s2W3, P23, 256, 512,  8,  32);
  pack_w_kernel<<<packGrid(17, 32), 256, 0, stream>>>(s3W1, P31, 515, 512,  17, 32);
  pack_w_kernel<<<packGrid(16, 32), 256, 0, stream>>>(s3W2, P32, 512, 512,  16, 32);
  pack_w_kernel<<<packGrid(16, 64), 256, 0, stream>>>(s3W3, P33, 512, 1024, 16, 64);

  // ---- SA1 ----
  fps_kernel<<<BB, 1024, 0, stream>>>(x, 6, NPTS, M1, idx1, ps1);
  ball_query_kernel<<<dim3((M1 + 127) / 128, BB), 128, (size_t)KNBR * 128 * 8, stream>>>(
      x, 6, ps1, NPTS, M1, 4.0f, nbr1, msk1);
  {
    size_t sm = (size_t)64 * 128 * 2 + (size_t)64 * 128 * 2 + 64 * 4 + 256;
    sa_conv_kernel<64, 32, 3, 128, 128, 256, false><<<dim3(M1, BB), 256, sm, stream>>>(
        x + 3, 6, x, 6, ps1, nbr1, msk1,
        P11, s1b1, P12, s1b2, P13, s1b3, x1, nullptr, NPTS, M1);
  }
  // ---- SA2 ----
  fps_kernel<<<BB, 1024, 0, stream>>>(ps1, 3, M1, M2, idx2, ps2);
  ball_query_kernel<<<dim3((M2 + 127) / 128, BB), 128, (size_t)KNBR * 128 * 8, stream>>>(
      ps1, 3, ps2, M1, M2, 16.0f, nbr2, msk2);
  {
    size_t sm = (size_t)64 * 288 * 2 + (size_t)64 * 256 * 2 + 64 * 4 + 256;
    sa_conv_kernel<64, 288, 256, 256, 256, 512, false><<<dim3(M2, BB), 256, sm, stream>>>(
        x1, 256, ps1, 3, ps2, nbr2, msk2,
        P21, s2b1, P22, s2b2, P23, s2b3, x2, nullptr, M1, M2);
  }
  // ---- SA3 (global) ----
  init_genc_kernel<<<(BB * 1024 + 255) / 256, 256, 0, stream>>>(genc, BB * 1024);
  {
    size_t sm = (size_t)32 * 544 * 2 + (size_t)32 * 512 * 2 + 32 * 4 + 256;
    sa_conv_kernel<32, 544, 512, 512, 512, 1024, true><<<dim3((M2 + 31) / 32, BB), 256, sm, stream>>>(
        x2, 512, ps2, 3, nullptr, nullptr, nullptr,
        P31, s3b1, P32, s3b2, P33, s3b3, nullptr, genc, M2, 0);
  }
  // ---- head + broadcast ----
  head_kernel<<<BB, 512, 0, stream>>>(genc, hW1, hb1, hW2, hb2, hW3, hb3, (float*)d_out);
}
